// SpatialConv_24008867185221
// MI455X (gfx1250) — matrix operand order repaired and checked by reading, not compiled, not run on hardware
//
#include <hip/hip_runtime.h>

// SpatialConv (Chebyshev graph conv with spatial attention) for MI455X / gfx1250.
// out[b,n,o,t] = relu( sum_k sum_m cheb[k,n,m]*attn[b,n,m] * sum_f x[b,m,f,t]*w[k,f,o] )
//
// Fused two-stage bf16 WMMA, one workgroup = (b, 64 rows of n), m in chunks of 32:
//   stage 1: Y_k[m, t*64+o]  = X[t][m,f] @ W_k[f,o]       (wmma bf16 -> LDS)
//   stage 2: acc[n, t*64+o] += A_k[n,m]  @ Y_k[m, t*64+o] (wmma bf16, regs)
// cheb/attn tiles are brought in by the Tensor Data Mover (tensor_load_to_lds),
// double-buffered and overlapped with compute; x is a transpose+convert fill.

typedef __attribute__((ext_vector_type(16))) __bf16 v16bf;
typedef __attribute__((ext_vector_type(8)))  float  v8f;
typedef __attribute__((ext_vector_type(4)))  float  f32x4;
typedef unsigned int u32;
typedef u32 __attribute__((ext_vector_type(4))) u32x4;
typedef int  __attribute__((ext_vector_type(4))) i32x4;
typedef int  __attribute__((ext_vector_type(8))) i32x8;

#ifndef __has_builtin
#define __has_builtin(x) 0
#endif
#if __has_builtin(__builtin_amdgcn_tensor_load_to_lds) && \
    __has_builtin(__builtin_amdgcn_s_wait_tensorcnt)
#define HAVE_TDM 1
#else
#define HAVE_TDM 0
#endif

namespace {
constexpr int Bt   = 32;
constexpr int Nn   = 512;
constexpr int FIN  = 64;
constexpr int Tt   = 12;
constexpr int Kc   = 3;
constexpr int FOUT = 64;
constexpr int TO   = Tt * FOUT;   // 768 fused (t,o) columns
constexpr int NB   = 64;          // n rows per workgroup
constexpr int MC   = 32;          // m chunk (one bf16 WMMA K-step)
constexpr int NCHUNK  = Nn / MC;  // 16
constexpr int THREADS = 1024;     // 32 wave32 waves

// LDS partition
constexpr int SY_ELEMS = Kc * TO * MC;     // 73728 bf16 : Y^T [k][to][m]
constexpr int SX_ELEMS = Tt * MC * FIN;    // 24576 bf16 : X   [t][m][f]
constexpr int SW_ELEMS = Kc * FOUT * FIN;  // 12288 bf16 : W^T [k][o][f]
constexpr int SA_ELEMS = Kc * NB * MC;     //  6144 bf16 : A   [k][n][m]
constexpr int STA_ELEMS = NB * MC;         //  2048 f32  : attn staging (per buf)
constexpr int STC_ELEMS = Kc * NB * MC;    //  6144 f32  : cheb staging (per buf)
constexpr size_t SMEM_BYTES =
    (size_t)(SY_ELEMS + SX_ELEMS + SW_ELEMS + SA_ELEMS) * sizeof(__bf16) +
    (size_t)(STA_ELEMS + STC_ELEMS) * 2 * sizeof(float);   // 299008 B
} // namespace

__device__ __forceinline__ __bf16 cvt_bf16(float x) {
  return (__bf16)x;   // native fptrunc; gfx1250 has hardware bf16 converts
}

// A-matrix 16x32 bf16 fragment (ISA 7.12.2): lane m = L&15; K runs
// {hi*8 .. hi*8+7} and {16+hi*8 .. 16+hi*8+7}. Rows contiguous in LDS.
__device__ __forceinline__ v16bf frag_a16x32(const __bf16* base, int rstride, int lane) {
  const int m = lane & 15, hi = lane >> 4;
  union { f32x4 q[2]; v16bf v; } u;
  const __bf16* p = base + m * rstride + hi * 8;
  u.q[0] = *(const f32x4*)(p);
  u.q[1] = *(const f32x4*)(p + 16);
  return u.v;
}

// B-matrix 32x16 bf16 fragment: lane col c = L&15; K runs {hi*8 .. hi*8+7} and {16+hi*8 .. 16+hi*8+7}.
// Columns stored contiguous-over-K in LDS.
__device__ __forceinline__ v16bf frag_b32x16(const __bf16* base, int cstride, int lane) {
  const int c = lane & 15, hi = lane >> 4;
  union { f32x4 q[2]; v16bf v; } u;
  const __bf16* p = base + c * cstride + hi * 8;
  u.q[0] = *(const f32x4*)(p);
  u.q[1] = *(const f32x4*)(p + 16);
  return u.v;
}

#if HAVE_TDM
// Issue one TDM load of an up-to-3D f32 tile into LDS (ISA §8.3-8.6).
//   tile (td0 x td1 x td2) elements, row stride s0, plane stride s1 (elements),
//   packed contiguously at lds_addr.
__device__ __forceinline__ void tdm_load_f32(u32 lds_addr, const void* gptr,
                                             u32 tile_d0, u32 tile_d1, u32 tile_d2,
                                             u32 tensor_d0, u32 tensor_d1, u32 tensor_d2,
                                             u32 s0, unsigned long long s1) {
  const unsigned long long ga = (unsigned long long)gptr;
  u32x4 g0;
  g0.x = 1u;                                               // count=1, user mode
  g0.y = lds_addr;                                         // LDS byte address
  g0.z = (u32)ga;                                          // global_addr[31:0]
  g0.w = (u32)((ga >> 32) & 0x01FFFFFFull) | (2u << 30);   // addr[56:32], type=2
  i32x8 g1;
  g1[0] = (int)(2u << 16);                                 // data_size = 4B
  g1[1] = (int)(tensor_d0 << 16);                          // tensor_dim0[15:0]
  g1[2] = (int)((tensor_d0 >> 16) | (tensor_d1 << 16));    // td0[31:16] | td1[15:0]
  g1[3] = (int)((tensor_d1 >> 16) | (tile_d0 << 16));      // td1[31:16] | tile_dim0
  g1[4] = (int)(tile_d1 | (tile_d2 << 16));                // tile_dim1 | tile_dim2
  g1[5] = (int)s0;                                         // dim0 stride [31:0]
  g1[6] = (int)((u32)(s1 & 0xffffull) << 16);              // s0[47:32]=0 | s1[15:0]
  g1[7] = (int)(s1 >> 16);                                 // s1[47:16]
  i32x4 g2;
  g2[0] = (int)tensor_d2;                                  // tensor_dim2
  g2[1] = 0; g2[2] = 0; g2[3] = 0;                         // dim3/stride2/tile_dim3
  i32x4 g3 = {0, 0, 0, 0};
#if defined(__clang_major__) && __clang_major__ >= 23
  i32x8 g4 = {0, 0, 0, 0, 0, 0, 0, 0};
  __builtin_amdgcn_tensor_load_to_lds(g0, g1, g2, g3, g4, 0);
#else
  __builtin_amdgcn_tensor_load_to_lds(g0, g1, g2, g3, 0);
#endif
}
#endif // HAVE_TDM

__global__ __launch_bounds__(THREADS, 1)
void spatial_cheb_wmma_kernel(const float* __restrict__ x,      // [B,N,FIN,T]
                              const float* __restrict__ attn,   // [B,N,N]
                              const float* __restrict__ cheb,   // [K,N,N]
                              const float* __restrict__ weight, // [K,FIN,FOUT]
                              float* __restrict__ out) {        // [B,N,FOUT,T]
  extern __shared__ __align__(16) char smem_raw[];
  __bf16* sY = (__bf16*)smem_raw;       // [Kc][TO][MC]  (m contiguous)
  __bf16* sX = sY + SY_ELEMS;           // [Tt][MC][FIN] (f contiguous)
  __bf16* sW = sX + SX_ELEMS;           // [Kc][FOUT][FIN] (f contiguous)
  __bf16* sA = sW + SW_ELEMS;           // [Kc][NB][MC]  (m contiguous)
  float*  sAttnS = (float*)(sA + SA_ELEMS); // [2][NB][MC]     TDM staging
  float*  sChebS = sAttnS + 2 * STA_ELEMS;  // [2][Kc][NB][MC] TDM staging

  const int tid  = threadIdx.x;
  const int lane = tid & 31;
  const int wave = tid >> 5;            // 0..31
  const int b    = blockIdx.y;
  const int n0   = blockIdx.x * NB;

  // One-time: weights -> LDS, transposed [k][o][f], bf16.
  for (int i = tid; i < SW_ELEMS; i += THREADS) {
    const int k = i / (FOUT * FIN);
    const int r = i - k * FOUT * FIN;
    const int o = r / FIN;
    const int f = r - o * FIN;
    sW[i] = cvt_bf16(weight[(k * FIN + f) * FOUT + o]);
  }

#if HAVE_TDM
  // Prologue: async-load chunk 0's attn/cheb tiles into staging buffer 0.
  if (wave == 0) {
    const u32 attn_lds = (u32)(uintptr_t)(void*)sAttnS;
    const u32 cheb_lds = (u32)(uintptr_t)(void*)sChebS;
    tdm_load_f32(attn_lds, attn + (size_t)(b * Nn + n0) * Nn,
                 MC, NB, 0, Nn, Nn, 0, Nn, 0ull);
    tdm_load_f32(cheb_lds, cheb + (size_t)n0 * Nn,
                 MC, NB, Kc, Nn, Nn, Kc, Nn, (unsigned long long)Nn * Nn);
  }
#endif

  // Persistent accumulators: wave -> (n subtile, group of 6 to-tiles).
  const int nsub = wave >> 3;           // 0..3 : n rows [nsub*16, nsub*16+16)
  const int jgrp = wave & 7;            // 0..7 : to-tiles [jgrp*6, jgrp*6+6)
  v8f acc[6];
  {
    const v8f z = {};
    #pragma unroll
    for (int jj = 0; jj < 6; ++jj) acc[jj] = z;
  }

  for (int mc = 0; mc < NCHUNK; ++mc) {
    const int m0  = mc * MC;
    const int cur = mc & 1;

#if HAVE_TDM
    if (wave == 0) __builtin_amdgcn_s_wait_tensorcnt(0);  // staging[cur] landed
#endif
    __syncthreads();  // staging visible to all; prior consumers done before refill

#if HAVE_TDM
    // Kick off next chunk's TDM into the other staging buffer (overlaps compute).
    if (wave == 0 && (mc + 1) < NCHUNK) {
      const int nxt = cur ^ 1, m1 = m0 + MC;
      const u32 attn_lds = (u32)(uintptr_t)(void*)(sAttnS + nxt * STA_ELEMS);
      const u32 cheb_lds = (u32)(uintptr_t)(void*)(sChebS + nxt * STC_ELEMS);
      tdm_load_f32(attn_lds, attn + (size_t)(b * Nn + n0) * Nn + m1,
                   MC, NB, 0, Nn, Nn, 0, Nn, 0ull);
      tdm_load_f32(cheb_lds, cheb + (size_t)n0 * Nn + m1,
                   MC, NB, Kc, Nn, Nn, Kc, Nn, (unsigned long long)Nn * Nn);
    }
#endif

    // Fill sX[t][m][f] <- x[b, m0+m, f, t] (t contiguous in global).
    for (int i = tid; i < SX_ELEMS; i += THREADS) {
      const int m = i / (FIN * Tt);
      const int r = i - m * FIN * Tt;
      const int f = r / Tt;
      const int t = r - f * Tt;
      sX[(t * MC + m) * FIN + f] =
          cvt_bf16(x[((b * Nn + (m0 + m)) * FIN + f) * Tt + t]);
    }

    // sA[k][n][m] <- cheb * attn, bf16.
#if HAVE_TDM
    {
      const float* stA = sAttnS + cur * STA_ELEMS;
      const float* stC = sChebS + cur * STC_ELEMS;
      for (int i = tid; i < SA_ELEMS; i += THREADS) {
        const int nm = i % (NB * MC);
        sA[i] = cvt_bf16(stC[i] * stA[nm]);
      }
    }
#else
    for (int i = tid; i < SA_ELEMS; i += THREADS) {
      const int k = i / (NB * MC);
      const int r = i - k * NB * MC;
      const int n = r / MC;
      const int m = r - n * MC;
      sA[i] = cvt_bf16(cheb[(k * Nn + (n0 + n)) * Nn + (m0 + m)] *
                       attn[(b * Nn + (n0 + n)) * Nn + (m0 + m)]);
    }
#endif

    // gfx1250: prefetch next m-chunk of x into L2 while we compute.
    if (mc + 1 < NCHUNK) {
      const float* nx = x + ((size_t)(b * Nn + (m0 + MC)) * FIN) * Tt;
      __builtin_prefetch(nx + (size_t)tid * 16, 0, 1);
    }
    __syncthreads();

    // ---- Stage 1: Y_k[m, t*64+o] = X[t][m,:] @ W_k[:,o], 288 C-tiles / chunk.
    for (int tt = wave; tt < Kc * Tt * 8; tt += 32) {
      const int k  = tt / 96;
      const int r  = tt - k * 96;
      const int t  = r >> 3;
      const int r2 = r & 7;
      const int ms = r2 >> 2;   // m subtile (0..1)
      const int ot = r2 & 3;    // o tile    (0..3)
      v8f c = {};
      #pragma unroll
      for (int f0 = 0; f0 < FIN; f0 += 32) {
        const v16bf a  = frag_a16x32(sX + (t * MC + ms * 16) * FIN + f0, FIN, lane);
        const v16bf bm = frag_b32x16(sW + (k * FOUT + ot * 16) * FIN + f0, FIN, lane);
        c = __builtin_amdgcn_wmma_f32_16x16x32_bf16(false, a, false, bm,
                                                    (short)0, c, false, false);
      }
      // Scatter C tile into sY^T ([k][to][m]) as bf16 for stage 2.
      const int col = lane & 15, hrow = (lane >> 4) << 3;
      const int to  = t * FOUT + ot * 16 + col;
      #pragma unroll
      for (int i2 = 0; i2 < 8; ++i2) {
        const int m = ms * 16 + i2 + hrow;
        sY[(k * TO + to) * MC + m] = cvt_bf16(c[i2]);
      }
    }
    __syncthreads();

    // ---- Stage 2: acc[n, to] += A_k[n, m-chunk] @ Y_k[m-chunk, to].
    #pragma unroll
    for (int k = 0; k < Kc; ++k) {
      const v16bf a = frag_a16x32(sA + (k * NB + nsub * 16) * MC, MC, lane);
      #pragma unroll
      for (int jj = 0; jj < 6; ++jj) {
        const int j = jgrp * 6 + jj;
        const v16bf bm = frag_b32x16(sY + (k * TO + j * 16) * MC, MC, lane);
        acc[jj] = __builtin_amdgcn_wmma_f32_16x16x32_bf16(false, a, false, bm,
                                                          (short)0, acc[jj], false, false);
      }
    }
  }

  // Epilogue: ReLU + store out[b][n][o][t].
  {
    const int col = lane & 15, hrow = (lane >> 4) << 3;
    #pragma unroll
    for (int jj = 0; jj < 6; ++jj) {
      const int j = jgrp * 6 + jj;
      const int t = j >> 2;               // 4 to-tiles per t
      const int o = (j & 3) * 16 + col;
      #pragma unroll
      for (int i2 = 0; i2 < 8; ++i2) {
        const int n = n0 + nsub * 16 + i2 + hrow;
        out[((b * Nn + n) * FOUT + o) * Tt + t] = fmaxf(acc[jj][i2], 0.0f);
      }
    }
  }
}

extern "C" void kernel_launch(void* const* d_in, const int* in_sizes, int n_in,
                              void* d_out, int out_size, void* d_ws, size_t ws_size,
                              hipStream_t stream) {
  (void)in_sizes; (void)n_in; (void)d_ws; (void)ws_size; (void)out_size;
  const float* x      = (const float*)d_in[0];
  const float* attn   = (const float*)d_in[1];
  const float* cheb   = (const float*)d_in[2];
  const float* weight = (const float*)d_in[3];
  float* out          = (float*)d_out;

  // ~292KB dynamic LDS (> default cap) — raise the limit. Deterministic,
  // idempotent host-side call; not a stream op, safe under graph capture.
  (void)hipFuncSetAttribute(reinterpret_cast<const void*>(spatial_cheb_wmma_kernel),
                            hipFuncAttributeMaxDynamicSharedMemorySize,
                            (int)SMEM_BYTES);

  dim3 grid(Nn / NB, Bt);   // 8 x 32 = 256 workgroups
  spatial_cheb_wmma_kernel<<<grid, dim3(THREADS), SMEM_BYTES, stream>>>(
      x, attn, cheb, weight, out);
}
